// hgnn_kpi2d_14422500180474
// MI455X (gfx1250) — matrix operand order repaired and checked by reading, not compiled, not run on hardware
//
#include <hip/hip_runtime.h>

#define HD   128
#define GNUM 256
#define ODIM 16

typedef __attribute__((ext_vector_type(16))) __bf16 v16bf;
typedef __attribute__((ext_vector_type(4)))  float  v4f;
typedef __attribute__((ext_vector_type(8)))  float  v8f;

// ---- A fragment: 16x32 bf16, row = lane&15, K-chunks selected by lane>>4 ----
// lanes 0-15 : K = k0+0..15
// lanes 16-31: K = k0+16..31
static __device__ __forceinline__ v16bf load_a_frag(const float* __restrict__ xrow,
                                                    int k0, int half, float rs) {
  const float* p = xrow + k0 + 16 * half;
  v4f a0 = *(const v4f*)(p);
  v4f a1 = *(const v4f*)(p + 4);
  v4f a2 = *(const v4f*)(p + 8);
  v4f a3 = *(const v4f*)(p + 12);
  v16bf a;
#pragma unroll
  for (int j = 0; j < 4; ++j) {
    a[j]      = (__bf16)(a0[j] * rs);
    a[4 + j]  = (__bf16)(a1[j] * rs);
    a[8 + j]  = (__bf16)(a2[j] * rs);
    a[12 + j] = (__bf16)(a3[j] * rs);
  }
  return a;
}

// ---- fused multi-term GEMM: Y[M,128] = sum_t diag(1/max(cnt_t,1)) X_t @ W_t  (+bias)*premul
// POOL: relu then atomicAdd into per-graph pool. Block = 256 thr = 8 waves, wave -> 16-col tile.
template <int NT, bool POOL>
__global__ __launch_bounds__(256) void wmma_gemm_fused(
    const float* __restrict__ X0, const float* __restrict__ c0, const __bf16* __restrict__ W0,
    const float* __restrict__ X1, const float* __restrict__ c1, const __bf16* __restrict__ W1,
    const float* __restrict__ X2, const float* __restrict__ c2, const __bf16* __restrict__ W2,
    const float* __restrict__ X3, const float* __restrict__ c3, const __bf16* __restrict__ W3,
    int K, const float* __restrict__ bias0, const float* __restrict__ bias1, float premul,
    float* __restrict__ Yout, float* __restrict__ pool, const int* __restrict__ batch) {
  const int lane = threadIdx.x & 31;
  const int wave = threadIdx.x >> 5;
  const int half = lane >> 4;
  const int l15  = lane & 15;
  const int rowA = blockIdx.x * 16 + l15;   // row this lane loads for A
  const int colg = wave * 16 + l15;         // output column (B col / C col)

  v8f acc = {0.f, 0.f, 0.f, 0.f, 0.f, 0.f, 0.f, 0.f};

  const float*  Xs[4] = {X0, X1, X2, X3};
  const float*  cs[4] = {c0, c1, c2, c3};
  const __bf16* Ws[4] = {W0, W1, W2, W3};

#pragma unroll
  for (int t = 0; t < NT; ++t) {
    float rs = 1.0f;
    if (cs[t]) rs = 1.0f / fmaxf(cs[t][rowA], 1.0f);   // fused mean-normalize
    const float*  xrow = Xs[t] + (size_t)rowA * K;
    const __bf16* wcol = Ws[t] + (size_t)colg * K;     // W pre-transposed: [128][K] bf16
    for (int k0 = 0; k0 < K; k0 += 32) {
      v16bf a = load_a_frag(xrow, k0, half, rs);
      v16bf b = *(const v16bf*)(wcol + k0 + 16 * half); // lanes0-15: K=k0..k0+15; 16-31: +16
      acc = __builtin_amdgcn_wmma_f32_16x16x32_bf16(false, a, false, b, (short)0, acc,
                                                    false, false);
    }
  }

  float bias = 0.f;
  if (bias0) bias += bias0[colg];
  if (bias1) bias += bias1[colg];
#pragma unroll
  for (int r = 0; r < 8; ++r) {               // C/D: VGPR r -> row r + 8*half
    int row = blockIdx.x * 16 + r + 8 * half;
    float v = (acc[r] + bias) * premul;
    if (POOL) {
      v = fmaxf(v, 0.f);                      // fused ReLU
      atomicAdd(pool + (size_t)batch[row] * HD + colg, v);   // fused graph pool
    } else {
      Yout[(size_t)row * HD + colg] = v;
    }
  }
}

// ---- edge aggregation: one wave per edge, float4 per lane, scatter-add in L2 ----
__global__ __launch_bounds__(256) void edge_aggregate(const int* __restrict__ ei, int E,
                                                      const float* __restrict__ hsrc,
                                                      float* __restrict__ agg,
                                                      float* __restrict__ cnt) {
  int gid  = blockIdx.x * 256 + threadIdx.x;
  int e    = gid >> 5;
  int lane = gid & 31;
  if (e >= E) return;
  int src = ei[e];
  int dst = ei[(size_t)E + e];
  v4f v = *(const v4f*)(hsrc + (size_t)src * HD + lane * 4);
  float* a = agg + (size_t)dst * HD + lane * 4;
  atomicAdd(a + 0, v[0]);
  atomicAdd(a + 1, v[1]);
  atomicAdd(a + 2, v[2]);
  atomicAdd(a + 3, v[3]);
  if (lane == 0) atomicAdd(cnt + dst, 1.0f);
}

__global__ void batch_count(const int* __restrict__ batch, int N, float* __restrict__ cnt) {
  int i = blockIdx.x * blockDim.x + threadIdx.x;
  if (i < N) atomicAdd(cnt + batch[i], 1.0f);
}

// transpose + convert weight: W[K][Nc] f32 -> Wt[Nc][K] bf16
__global__ void prep_weight(const float* __restrict__ W, int K, int Nc,
                            __bf16* __restrict__ Wt) {
  int i = blockIdx.x * blockDim.x + threadIdx.x;
  if (i >= K * Nc) return;
  int n = i / K, k = i - n * K;
  Wt[(size_t)n * K + k] = (__bf16)W[(size_t)k * Nc + n];
}

// xcomb = 0.5*(pool_a/max(cnt_a,1) + pool_b/max(cnt_b,1))
__global__ void finalize_pool(const float* __restrict__ pa, const float* __restrict__ ca,
                              const float* __restrict__ pb, const float* __restrict__ cb,
                              float* __restrict__ xcomb) {
  int i = blockIdx.x * blockDim.x + threadIdx.x;  // < GNUM*HD
  int g = i >> 7;
  float va = pa[i] / fmaxf(ca[g], 1.0f);
  float vb = pb[i] / fmaxf(cb[g], 1.0f);
  xcomb[i] = 0.5f * (va + vb);
}

// out[G,16] = xcomb @ Wout + bout ; one wave per 16-row tile, single 16-col tile
__global__ __launch_bounds__(32) void wmma_out_head(const float* __restrict__ X,
                                                    const __bf16* __restrict__ Wt,
                                                    const float* __restrict__ bout,
                                                    float* __restrict__ out) {
  int lane = threadIdx.x & 31;
  int half = lane >> 4, l15 = lane & 15;
  int rowA = blockIdx.x * 16 + l15;
  v8f acc = {0.f, 0.f, 0.f, 0.f, 0.f, 0.f, 0.f, 0.f};
  const float*  xrow = X + (size_t)rowA * HD;
  const __bf16* wcol = Wt + (size_t)l15 * HD;
  for (int k0 = 0; k0 < HD; k0 += 32) {
    v16bf a = load_a_frag(xrow, k0, half, 1.0f);
    v16bf b = *(const v16bf*)(wcol + k0 + 16 * half);
    acc = __builtin_amdgcn_wmma_f32_16x16x32_bf16(false, a, false, b, (short)0, acc,
                                                  false, false);
  }
  float bias = bout[l15];
#pragma unroll
  for (int r = 0; r < 8; ++r) {
    int row = blockIdx.x * 16 + r + 8 * half;
    out[row * ODIM + l15] = acc[r] + bias;
  }
}

extern "C" void kernel_launch(void* const* d_in, const int* in_sizes, int n_in,
                              void* d_out, int out_size, void* d_ws, size_t ws_size,
                              hipStream_t stream) {
  const int N_A = 100000, N_B = 100000, E = 1600000;

  const float* x_a     = (const float*)d_in[0];
  const float* x_b     = (const float*)d_in[1];
  const int*   ei_ab   = (const int*)d_in[2];
  const int*   ei_ba   = (const int*)d_in[3];
  const int*   ei_aa   = (const int*)d_in[4];
  const int*   batch_a = (const int*)d_in[5];
  const int*   batch_b = (const int*)d_in[6];
  const float* Wemb_a  = (const float*)d_in[7];
  const float* bemb_a  = (const float*)d_in[8];
  const float* Wemb_b  = (const float*)d_in[9];
  const float* bemb_b  = (const float*)d_in[10];
  const float* Wl_ab   = (const float*)d_in[11];
  const float* bl_ab   = (const float*)d_in[12];
  const float* Wr_ab   = (const float*)d_in[13];
  const float* Wl_ba   = (const float*)d_in[14];
  const float* bl_ba   = (const float*)d_in[15];
  const float* Wr_ba   = (const float*)d_in[16];
  const float* Wl_aa   = (const float*)d_in[17];
  const float* bl_aa   = (const float*)d_in[18];
  const float* Wr_aa   = (const float*)d_in[19];
  const float* Wout    = (const float*)d_in[20];
  const float* bout    = (const float*)d_in[21];

  // ---- workspace layout (all 32B aligned by construction) ----
  float* ws = (float*)d_ws;
  size_t off = 0;
  float* h_a    = ws + off; off += (size_t)N_A * HD;
  float* h_b    = ws + off; off += (size_t)N_B * HD;
  float* zbase  = ws + off;                       // start of zero-init region
  float* agg_ab = ws + off; off += (size_t)N_B * HD;
  float* agg_ba = ws + off; off += (size_t)N_A * HD;
  float* agg_aa = ws + off; off += (size_t)N_A * HD;
  float* cnt_ab = ws + off; off += N_B;
  float* cnt_ba = ws + off; off += N_A;
  float* cnt_aa = ws + off; off += N_A;
  float* pool_a = ws + off; off += (size_t)GNUM * HD;
  float* pool_b = ws + off; off += (size_t)GNUM * HD;
  float* gca    = ws + off; off += GNUM;
  float* gcb    = ws + off; off += GNUM;
  size_t zero_floats = (size_t)(ws + off - zbase);
  float* xcomb  = ws + off; off += (size_t)GNUM * HD;

  __bf16* wt = (__bf16*)(ws + off);
  size_t wo = 0;
  __bf16* wt_emb_a = wt + wo; wo += (size_t)HD * 64;
  __bf16* wt_emb_b = wt + wo; wo += (size_t)HD * 32;
  __bf16* wt_l_ab  = wt + wo; wo += (size_t)HD * HD;
  __bf16* wt_r_ab  = wt + wo; wo += (size_t)HD * HD;
  __bf16* wt_l_ba  = wt + wo; wo += (size_t)HD * HD;
  __bf16* wt_r_ba  = wt + wo; wo += (size_t)HD * HD;
  __bf16* wt_l_aa  = wt + wo; wo += (size_t)HD * HD;
  __bf16* wt_r_aa  = wt + wo; wo += (size_t)HD * HD;
  __bf16* wt_out   = wt + wo; wo += (size_t)ODIM * HD;

  hipMemsetAsync(zbase, 0, zero_floats * sizeof(float), stream);

  // weight prep (transpose + f32->bf16)
  prep_weight<<<(64 * HD + 255) / 256, 256, 0, stream>>>(Wemb_a, 64, HD, wt_emb_a);
  prep_weight<<<(32 * HD + 255) / 256, 256, 0, stream>>>(Wemb_b, 32, HD, wt_emb_b);
  prep_weight<<<(HD * HD + 255) / 256, 256, 0, stream>>>(Wl_ab, HD, HD, wt_l_ab);
  prep_weight<<<(HD * HD + 255) / 256, 256, 0, stream>>>(Wr_ab, HD, HD, wt_r_ab);
  prep_weight<<<(HD * HD + 255) / 256, 256, 0, stream>>>(Wl_ba, HD, HD, wt_l_ba);
  prep_weight<<<(HD * HD + 255) / 256, 256, 0, stream>>>(Wr_ba, HD, HD, wt_r_ba);
  prep_weight<<<(HD * HD + 255) / 256, 256, 0, stream>>>(Wl_aa, HD, HD, wt_l_aa);
  prep_weight<<<(HD * HD + 255) / 256, 256, 0, stream>>>(Wr_aa, HD, HD, wt_r_aa);
  prep_weight<<<(HD * ODIM + 255) / 256, 256, 0, stream>>>(Wout, HD, ODIM, wt_out);

  // per-type input embeddings (WMMA, store f32)
  wmma_gemm_fused<1, false><<<N_A / 16, 256, 0, stream>>>(
      x_a, nullptr, wt_emb_a, nullptr, nullptr, nullptr, nullptr, nullptr, nullptr,
      nullptr, nullptr, nullptr, 64, bemb_a, nullptr, 1.0f, h_a, nullptr, nullptr);
  wmma_gemm_fused<1, false><<<N_B / 16, 256, 0, stream>>>(
      x_b, nullptr, wt_emb_b, nullptr, nullptr, nullptr, nullptr, nullptr, nullptr,
      nullptr, nullptr, nullptr, 32, bemb_b, nullptr, 1.0f, h_b, nullptr, nullptr);

  // edge aggregation (gather + L2 scatter-add), one wave per edge
  int eblocks = (E * 32 + 255) / 256;
  edge_aggregate<<<eblocks, 256, 0, stream>>>(ei_ab, E, h_a, agg_ab, cnt_ab);
  edge_aggregate<<<eblocks, 256, 0, stream>>>(ei_ba, E, h_b, agg_ba, cnt_ba);
  edge_aggregate<<<eblocks, 256, 0, stream>>>(ei_aa, E, h_a, agg_aa, cnt_aa);

  // per-graph node counts
  batch_count<<<(N_A + 255) / 256, 256, 0, stream>>>(batch_a, N_A, gca);
  batch_count<<<(N_B + 255) / 256, 256, 0, stream>>>(batch_b, N_B, gcb);

  // dst 'b': agg_ab@Wl_ab + h_b@Wr_ab + bl_ab -> relu -> pool_b
  wmma_gemm_fused<2, true><<<N_B / 16, 256, 0, stream>>>(
      agg_ab, cnt_ab, wt_l_ab, h_b, nullptr, wt_r_ab, nullptr, nullptr, nullptr,
      nullptr, nullptr, nullptr, HD, bl_ab, nullptr, 1.0f, nullptr, pool_b, batch_b);
  // dst 'a': 0.5*(agg_ba@Wl_ba + agg_aa@Wl_aa + h_a@Wr_ba + h_a@Wr_aa + bl_ba + bl_aa)
  wmma_gemm_fused<4, true><<<N_A / 16, 256, 0, stream>>>(
      agg_ba, cnt_ba, wt_l_ba, agg_aa, cnt_aa, wt_l_aa, h_a, nullptr, wt_r_ba,
      h_a, nullptr, wt_r_aa, HD, bl_ba, bl_aa, 0.5f, nullptr, pool_a, batch_a);

  // combine pools, output head
  finalize_pool<<<(GNUM * HD) / 256, 256, 0, stream>>>(pool_a, gca, pool_b, gcb, xcomb);
  wmma_out_head<<<GNUM / 16, 32, 0, stream>>>(xcomb, wt_out, bout, (float*)d_out);

  (void)in_sizes; (void)n_in; (void)out_size; (void)ws_size;
}
